// MultiHeadSelfAttention_1855425872285
// MI455X (gfx1250) — compile-verified, matrix operand order checked by reading, not run on hardware
//
#include <hip/hip_runtime.h>

typedef __attribute__((ext_vector_type(16))) _Float16 v16h;
typedef __attribute__((ext_vector_type(8)))  float    v8f;
typedef __attribute__((ext_vector_type(2)))  _Float16 h2;

#define B_  4
#define S_  2048
#define E_  1024
#define H_  16
#define DH_ 64

// k offset within a 16-element (8 VGPR) wave32 WMMA fragment, per ISA 7.12.2:
// VGPR vg holds k = kbase + {0,2,4,6,16,18,20,22}[vg] (+0/+1 for the two halves)
__device__ __forceinline__ int kl_of(int vg, int kbase) {
  return kbase + ((vg & 3) << 1) + ((vg >> 2) << 4);
}

__device__ __forceinline__ v8f wmma_f16(v16h a, v16h b, v8f c) {
  return __builtin_amdgcn_wmma_f32_16x16x32_f16(false, a, false, b, (short)0, c,
                                                false, false);
}

// xor-butterfly within each row of 16 lanes via v_permlane16_b32 (pure VALU)
template <unsigned LO, unsigned HI>
__device__ __forceinline__ float perml16(float x) {
  const int i = __builtin_bit_cast(int, x);
  const int y =
      __builtin_amdgcn_permlane16(i, i, (int)LO, (int)HI, false, false);
  return __builtin_bit_cast(float, y);
}
__device__ __forceinline__ float rowmax16(float x) {
  x = fmaxf(x, perml16<0x67452301u, 0xEFCDAB89u>(x));  // xor 1
  x = fmaxf(x, perml16<0x54761032u, 0xDCFE98BAu>(x));  // xor 2
  x = fmaxf(x, perml16<0x32107654u, 0xBA98FEDCu>(x));  // xor 4
  x = fmaxf(x, perml16<0xFEDCBA98u, 0x76543210u>(x));  // xor 8
  return x;
}

// ---------------- weight f32 -> f16 conversion ----------------
__global__ __launch_bounds__(256) void cvt_f16(const float* __restrict__ src,
                                               _Float16* __restrict__ dst) {
  const int i = (blockIdx.x * 256 + threadIdx.x) * 4;
  const float4 f = *(const float4*)(src + i);
  dst[i + 0] = (_Float16)f.x;
  dst[i + 1] = (_Float16)f.y;
  dst[i + 2] = (_Float16)f.z;
  dst[i + 3] = (_Float16)f.w;
}

// ---------------- WMMA GEMM: Out = (A(8192x1024)*W(1024x1024)+bias)*scale ---
// Block tile 128(M) x 64(N), k-step 32. 8 waves; wave (wm,wn) computes a
// 64x16 strip as 4 stacked 16x16 WMMA tiles.
template <bool A_F16, bool OUT_HEADS>
__global__ __launch_bounds__(256) void gemm_wmma(const void* __restrict__ Ap,
                                                 const _Float16* __restrict__ W,
                                                 const float* __restrict__ bias,
                                                 void* __restrict__ Out,
                                                 float out_scale) {
  __shared__ _Float16 sA[128 * 40];  // [row][k], padded stride 40
  __shared__ _Float16 sBt[64 * 40];  // transposed: [n][k], padded stride 40

  const int t = threadIdx.x, wid = t >> 5, lane = t & 31;
  const int lane16 = lane & 15, kbase = (lane < 16) ? 0 : 8;
  const int wm = wid & 1, wn = wid >> 1;
  const int blockM = blockIdx.x * 128, blockN = blockIdx.y * 64;

  v8f acc[4];
#pragma unroll
  for (int i = 0; i < 4; ++i)
    acc[i] = (v8f){0.f, 0.f, 0.f, 0.f, 0.f, 0.f, 0.f, 0.f};

  for (int k0 = 0; k0 < E_; k0 += 32) {
    __syncthreads();
    {  // A tile: 128x32, each thread does half a row (16 elems)
      const int row = t >> 1, half = t & 1;
      const size_t g = (size_t)(blockM + row) * E_ + k0 + half * 16;
      _Float16* dst = &sA[row * 40 + half * 16];
      if (A_F16) {
        const uint4* s = (const uint4*)((const _Float16*)Ap + g);
        ((uint4*)dst)[0] = s[0];
        ((uint4*)dst)[1] = s[1];
      } else {
        const float4* s = (const float4*)((const float*)Ap + g);
        const float4 f0 = s[0], f1 = s[1], f2 = s[2], f3 = s[3];
        _Float16 tmp[16] __attribute__((aligned(16)));
        tmp[0] = (_Float16)f0.x;  tmp[1] = (_Float16)f0.y;
        tmp[2] = (_Float16)f0.z;  tmp[3] = (_Float16)f0.w;
        tmp[4] = (_Float16)f1.x;  tmp[5] = (_Float16)f1.y;
        tmp[6] = (_Float16)f1.z;  tmp[7] = (_Float16)f1.w;
        tmp[8] = (_Float16)f2.x;  tmp[9] = (_Float16)f2.y;
        tmp[10] = (_Float16)f2.z; tmp[11] = (_Float16)f2.w;
        tmp[12] = (_Float16)f3.x; tmp[13] = (_Float16)f3.y;
        tmp[14] = (_Float16)f3.z; tmp[15] = (_Float16)f3.w;
        ((uint4*)dst)[0] = ((const uint4*)tmp)[0];
        ((uint4*)dst)[1] = ((const uint4*)tmp)[1];
      }
    }
    {  // B tile: 32x64 from W, stored transposed [n][k]
      const int kr = t >> 3, c8 = (t & 7) * 8;
      const uint4 u = *(const uint4*)(W + (size_t)(k0 + kr) * E_ + blockN + c8);
      _Float16 wv8[8] __attribute__((aligned(16)));
      *(uint4*)wv8 = u;
#pragma unroll
      for (int j = 0; j < 8; ++j) sBt[(c8 + j) * 40 + kr] = wv8[j];
    }
    __syncthreads();

    v16h bf;
#pragma unroll
    for (int vg = 0; vg < 8; ++vg) {
      const h2 p = *(const h2*)&sBt[(wn * 16 + lane16) * 40 + kl_of(vg, kbase)];
      bf[2 * vg] = p.x;
      bf[2 * vg + 1] = p.y;
    }
#pragma unroll
    for (int i = 0; i < 4; ++i) {
      v16h af;
      const int rb = wm * 64 + i * 16 + lane16;
#pragma unroll
      for (int vg = 0; vg < 8; ++vg) {
        const h2 p = *(const h2*)&sA[rb * 40 + kl_of(vg, kbase)];
        af[2 * vg] = p.x;
        af[2 * vg + 1] = p.y;
      }
      acc[i] = wmma_f16(af, bf, acc[i]);
    }
  }

  const int gcol = blockN + wn * 16 + lane16;
  const float bvv = bias[gcol];
#pragma unroll
  for (int i = 0; i < 4; ++i) {
#pragma unroll
    for (int r = 0; r < 8; ++r) {
      const int grow = blockM + wm * 64 + i * 16 + ((lane < 16) ? r : r + 8);
      const float val = (acc[i][r] + bvv) * out_scale;
      if (OUT_HEADS) {  // scatter into [B,H,S,DH] f16
        const int hh = gcol >> 6, d = gcol & 63;
        const int bb = grow >> 11, ss = grow & (S_ - 1);
        ((_Float16*)Out)[(((size_t)bb * H_ + hh) * S_ + ss) * DH_ + d] =
            (_Float16)val;
      } else {
        ((float*)Out)[(size_t)grow * E_ + gcol] = val;
      }
    }
  }
}

// ---------------- flash attention: per (b,h), 128 q-rows per WG -------------
// Q is pre-scaled by log2(e)/sqrt(DH) in the Q projection epilogue.
__global__ __launch_bounds__(256) void attn_wmma(const _Float16* __restrict__ Q,
                                                 const _Float16* __restrict__ K,
                                                 const _Float16* __restrict__ V,
                                                 _Float16* __restrict__ Cc) {
  __shared__ _Float16 sK[64 * 72];       // [key][d]
  __shared__ _Float16 sVt[64 * 72];      // transposed [d][key]
  __shared__ _Float16 sP[8 * 16 * 72];   // per-wave P staging [m][key]

  const int t = threadIdx.x, wid = t >> 5, lane = t & 31;
  const int lane16 = lane & 15, kbase = (lane < 16) ? 0 : 8;
  const int h = blockIdx.y, b = blockIdx.z;
  const int q0 = blockIdx.x * 128 + wid * 16;
  const size_t hb = ((size_t)b * H_ + h) * S_ * DH_;
  const _Float16* Qh = Q + hb;
  const _Float16* Kh = K + hb;
  const _Float16* Vh = V + hb;

  // Q fragments (16 rows x 64 d) as two 16x32 A-fragments, kept in registers
  v16h qa[2];
  {
    const int qrow = q0 + lane16;
#pragma unroll
    for (int ds = 0; ds < 2; ++ds)
#pragma unroll
      for (int vg = 0; vg < 8; ++vg) {
        const h2 p =
            *(const h2*)&Qh[(size_t)qrow * DH_ + ds * 32 + kl_of(vg, kbase)];
        qa[ds][2 * vg] = p.x;
        qa[ds][2 * vg + 1] = p.y;
      }
  }

  // constant all-ones B fragment: row sums of P via WMMA (P . 1)
  v16h onesb;
#pragma unroll
  for (int e = 0; e < 16; ++e) onesb[e] = (_Float16)1.0f;

  v8f o[4];
#pragma unroll
  for (int i = 0; i < 4; ++i)
    o[i] = (v8f){0.f, 0.f, 0.f, 0.f, 0.f, 0.f, 0.f, 0.f};
  float mrun[8], lrun[8];
#pragma unroll
  for (int r = 0; r < 8; ++r) { mrun[r] = -1e30f; lrun[r] = 0.f; }

  _Float16* myP = &sP[wid * 16 * 72];

  // per-thread staging addresses (64 keys x 64 d tile; 16 halves per thread)
  const int skey = t >> 2, sd0 = (t & 3) * 16;
  const unsigned ldsK =
      (unsigned)(uintptr_t)&sK[skey * 72 + sd0];  // low 32 bits = LDS offset

  for (int kb = 0; kb < S_ / 64; ++kb) {
    const int key0 = kb * 64;
    __syncthreads();
    {
      // K tile: async global -> LDS (32 bytes per thread, 2 x b128),
      // tracked by ASYNCcnt
      const _Float16* kg = &Kh[(size_t)(key0 + skey) * DH_ + sd0];
      asm volatile("global_load_async_to_lds_b128 %0, %1, off"
                   :
                   : "v"(ldsK), "v"(kg)
                   : "memory");
      asm volatile("global_load_async_to_lds_b128 %0, %1, off offset:16"
                   :
                   : "v"(ldsK), "v"(kg)
                   : "memory");
      // V tile: load to regs, store transposed [d][key]
      const _Float16* vg_ = &Vh[(size_t)(key0 + skey) * DH_ + sd0];
      _Float16 tv[16] __attribute__((aligned(16)));
      ((uint4*)tv)[0] = ((const uint4*)vg_)[0];
      ((uint4*)tv)[1] = ((const uint4*)vg_)[1];
#pragma unroll
      for (int j = 0; j < 16; ++j) sVt[(sd0 + j) * 72 + skey] = tv[j];
      asm volatile("s_wait_asynccnt 0x0" ::: "memory");
    }
    __syncthreads();

    // scores S = Q * K^T : 4 tiles of 16 keys, K-dim 64 = 2 x 32
    v8f sc[4];
#pragma unroll
    for (int kt = 0; kt < 4; ++kt) {
      sc[kt] = (v8f){0.f, 0.f, 0.f, 0.f, 0.f, 0.f, 0.f, 0.f};
#pragma unroll
      for (int ds = 0; ds < 2; ++ds) {
        v16h bf;
#pragma unroll
        for (int vg = 0; vg < 8; ++vg) {
          const h2 p = *(const h2*)&sK[(kt * 16 + lane16) * 72 + ds * 32 +
                                       kl_of(vg, kbase)];
          bf[2 * vg] = p.x;
          bf[2 * vg + 1] = p.y;
        }
        sc[kt] = wmma_f16(qa[ds], bf, sc[kt]);
      }
    }

    // online softmax max update; row m lives in VGPR r across lanes 0-15
    // (m=r) / lanes 16-31 (m=r+8); reduce within 16-lane rows via permlane
#pragma unroll
    for (int r = 0; r < 8; ++r) {
      float mx = fmaxf(fmaxf(sc[0][r], sc[1][r]), fmaxf(sc[2][r], sc[3][r]));
      mx = rowmax16(mx);
      const float mnew = fmaxf(mrun[r], mx);
      const float al = exp2f(mrun[r] - mnew);
      mrun[r] = mnew;
      lrun[r] *= al;
#pragma unroll
      for (int dt = 0; dt < 4; ++dt) o[dt][r] *= al;
#pragma unroll
      for (int kt = 0; kt < 4; ++kt)
        sc[kt][r] = exp2f(sc[kt][r] - mnew);
    }

    // C-layout -> A-fragment layout via private LDS patch (wave-local,
    // LDS ops from one wave stay in order; compiler barrier keeps order)
#pragma unroll
    for (int kt = 0; kt < 4; ++kt)
#pragma unroll
      for (int r = 0; r < 8; ++r) {
        const int m = (lane < 16) ? r : r + 8;
        myP[m * 72 + kt * 16 + lane16] = (_Float16)sc[kt][r];
      }
    __asm__ volatile("" ::: "memory");

    // O += P * V (4 d-tiles) and row-sums l += P * 1 (ones fragment),
    // K-dim 64 = 2 x 32
    v8f lacc = (v8f){0.f, 0.f, 0.f, 0.f, 0.f, 0.f, 0.f, 0.f};
#pragma unroll
    for (int ks = 0; ks < 2; ++ks) {
      v16h pa;
#pragma unroll
      for (int vg = 0; vg < 8; ++vg) {
        const h2 p = *(const h2*)&myP[lane16 * 72 + ks * 32 + kl_of(vg, kbase)];
        pa[2 * vg] = p.x;
        pa[2 * vg + 1] = p.y;
      }
      lacc = wmma_f16(pa, onesb, lacc);
#pragma unroll
      for (int dt = 0; dt < 4; ++dt) {
        v16h vb;
#pragma unroll
        for (int vg = 0; vg < 8; ++vg) {
          const h2 q = *(const h2*)&sVt[(dt * 16 + lane16) * 72 + ks * 32 +
                                        kl_of(vg, kbase)];
          vb[2 * vg] = q.x;
          vb[2 * vg + 1] = q.y;
        }
        o[dt] = wmma_f16(pa, vb, o[dt]);
      }
    }
#pragma unroll
    for (int r = 0; r < 8; ++r) lrun[r] += lacc[r];
  }

  // normalize and write concat [B*S, E] as f16
#pragma unroll
  for (int r = 0; r < 8; ++r) {
    const float inv = 1.0f / lrun[r];
    const int m = (lane < 16) ? r : r + 8;
    const size_t grow = (size_t)b * S_ + q0 + m;
#pragma unroll
    for (int dt = 0; dt < 4; ++dt) {
      const int gcol = h * DH_ + dt * 16 + lane16;
      Cc[grow * E_ + gcol] = (_Float16)(o[dt][r] * inv);
    }
  }
}

// ---------------------------------------------------------------------------
extern "C" void kernel_launch(void* const* d_in, const int* in_sizes, int n_in,
                              void* d_out, int out_size, void* d_ws,
                              size_t ws_size, hipStream_t stream) {
  (void)in_sizes; (void)n_in; (void)out_size; (void)ws_size;
  const float* x  = (const float*)d_in[0];
  // d_in[1] = mask (unused, faithful to reference)
  const float* wq = (const float*)d_in[2];
  const float* bq = (const float*)d_in[3];
  const float* wk = (const float*)d_in[4];
  const float* bk = (const float*)d_in[5];
  const float* wv = (const float*)d_in[6];
  const float* bv = (const float*)d_in[7];
  const float* wo = (const float*)d_in[8];
  const float* bo = (const float*)d_in[9];

  _Float16* p = (_Float16*)d_ws;
  _Float16* wq16 = p; p += (size_t)E_ * E_;
  _Float16* wk16 = p; p += (size_t)E_ * E_;
  _Float16* wv16 = p; p += (size_t)E_ * E_;
  _Float16* wo16 = p; p += (size_t)E_ * E_;
  _Float16* Q16  = p; p += (size_t)B_ * S_ * E_;
  _Float16* K16  = p; p += (size_t)B_ * S_ * E_;
  _Float16* V16  = p; p += (size_t)B_ * S_ * E_;
  _Float16* C16  = p;

  const int cvtGrid = (E_ * E_) / (256 * 4);
  cvt_f16<<<cvtGrid, 256, 0, stream>>>(wq, wq16);
  cvt_f16<<<cvtGrid, 256, 0, stream>>>(wk, wk16);
  cvt_f16<<<cvtGrid, 256, 0, stream>>>(wv, wv16);
  cvt_f16<<<cvtGrid, 256, 0, stream>>>(wv, wv16);
  cvt_f16<<<cvtGrid, 256, 0, stream>>>(wo, wo16);

  // softmax scale log2(e)/sqrt(DH) folded into the Q projection
  const float SC = 0.18033688011112042f;

  dim3 gg((B_ * S_) / 128, E_ / 64);
  gemm_wmma<false, true><<<gg, dim3(256), 0, stream>>>(x, wq16, bq, Q16, SC);
  gemm_wmma<false, true><<<gg, dim3(256), 0, stream>>>(x, wk16, bk, K16, 1.0f);
  gemm_wmma<false, true><<<gg, dim3(256), 0, stream>>>(x, wv16, bv, V16, 1.0f);

  attn_wmma<<<dim3(S_ / 128, H_, B_), dim3(256), 0, stream>>>(Q16, K16, V16,
                                                              C16);

  gemm_wmma<true, false><<<gg, dim3(256), 0, stream>>>(C16, wo16, bo,
                                                       (float*)d_out, 1.0f);
}
